// Attention_15040975471032
// MI455X (gfx1250) — matrix operand order repaired and checked by reading, not compiled, not run on hardware
//
#include <hip/hip_runtime.h>
#include <stdint.h>

typedef __attribute__((ext_vector_type(16))) __bf16 v16bf;
typedef __attribute__((ext_vector_type(8)))  float  v8f;
typedef unsigned short u16;

#define B_  4
#define N_  2048
#define C_  768
#define H_  12
#define D_  64
#define SCALE_ 0.125f   /* 64^-0.5 */

static __device__ __forceinline__ u16 f2bf(float f) {
  union { float f; uint32_t u; } x; x.f = f;
  uint32_t u = x.u;
  uint32_t r = (u + 0x7FFFu + ((u >> 16) & 1u)) >> 16;  // round-to-nearest-even
  return (u16)r;
}

static __device__ __forceinline__ v8f vzero8() {
  v8f r;
#pragma unroll
  for (int i = 0; i < 8; ++i) r[i] = 0.0f;
  return r;
}

// A fragment: 16x32 bf16, row-major source with stride ld (elements).
// Lane holds row m = lane&15; VGPR i holds packed K pair per ISA layout:
//   i<4 : K = 8*half + 2i      ; i>=4 : K = 16 + 8*half + 2(i-4)
static __device__ __forceinline__ v16bf load_a_frag(const u16* p, int ld, int lane) {
  int m = lane & 15, half = lane >> 4;
  union { uint32_t u[8]; v16bf v; } f;
#pragma unroll
  for (int i = 0; i < 8; ++i) {
    int k = (i < 4) ? (half * 8 + 2 * i) : (16 + half * 8 + 2 * (i - 4));
    f.u[i] = *(const uint32_t*)(p + (size_t)m * ld + k);
  }
  return f.v;
}

// B fragment: 32x16 bf16, B[k][n] = src[n*ld + k] (pairs contiguous along k).
// Lane holds column n = lane&15; VGPR i: i<4 : K = 8*half + 2i ; i>=4 : K = 16 + 8*half + 2(i-4).
static __device__ __forceinline__ v16bf load_b_frag(const u16* p, int ld, int lane) {
  int n = lane & 15, half = lane >> 4;
  union { uint32_t u[8]; v16bf v; } f;
#pragma unroll
  for (int i = 0; i < 8; ++i) {
    int k = (i < 4) ? (half * 8 + 2 * i) : (16 + half * 8 + 2 * (i - 4));
    f.u[i] = *(const uint32_t*)(p + (size_t)n * ld + k);
  }
  return f.v;
}

static __device__ __forceinline__ v8f wmma_bf16(v16bf a, v16bf b, v8f c) {
  return __builtin_amdgcn_wmma_f32_16x16x32_bf16(false, a, false, b, (short)0, c,
                                                 false, false);
}

// ---------------------------------------------------------------- conversion
__global__ void cvt_f32_bf16(const float* __restrict__ src, u16* __restrict__ dst, int n) {
  int i = blockIdx.x * blockDim.x + threadIdx.x;
  if (i < n) dst[i] = f2bf(src[i]);
}

// ------------------------------------------------------------- QKV GEMM
// Out[r][o] = sum_c X[r][c] * Wqkv[o][c]; one wave -> 32 rows x 64 cols,
// software-pipelined (fragments for step c+32 loaded while step c computes).
// Scatter: o -> (s,h,d); s=0 -> Q*SCALE [b,h,n,d]; s=1 -> K [b,h,n,d];
// s=2 -> Vt [b,h,d,n] (transposed for the PV WMMA B-fragment loads).
__global__ __launch_bounds__(32)
void qkv_gemm_kernel(const u16* __restrict__ xb, const u16* __restrict__ wb,
                     u16* __restrict__ Qb, u16* __restrict__ Kb, u16* __restrict__ Vtb) {
  int lane = threadIdx.x;
  int o0 = blockIdx.x * 64;
  int r0 = blockIdx.y * 32;
  v8f acc[2][4];
#pragma unroll
  for (int rb = 0; rb < 2; ++rb)
#pragma unroll
    for (int t = 0; t < 4; ++t) acc[rb][t] = vzero8();

  v16bf aC[2], bC[4];
#pragma unroll
  for (int rb = 0; rb < 2; ++rb)
    aC[rb] = load_a_frag(xb + (size_t)(r0 + rb * 16) * C_, C_, lane);
#pragma unroll
  for (int t = 0; t < 4; ++t)
    bC[t] = load_b_frag(wb + (size_t)(o0 + t * 16) * C_, C_, lane);

  for (int c = 0; c < C_; c += 32) {
    int cn = (c + 32 < C_) ? (c + 32) : 0;   // wrap: harmless extra load at tail
    v16bf aN[2], bN[4];
#pragma unroll
    for (int rb = 0; rb < 2; ++rb)
      aN[rb] = load_a_frag(xb + (size_t)(r0 + rb * 16) * C_ + cn, C_, lane);
#pragma unroll
    for (int t = 0; t < 4; ++t)
      bN[t] = load_b_frag(wb + (size_t)(o0 + t * 16) * C_ + cn, C_, lane);
#pragma unroll
    for (int rb = 0; rb < 2; ++rb)
#pragma unroll
      for (int t = 0; t < 4; ++t)
        acc[rb][t] = wmma_bf16(aC[rb], bC[t], acc[rb][t]);
#pragma unroll
    for (int rb = 0; rb < 2; ++rb) aC[rb] = aN[rb];
#pragma unroll
    for (int t = 0; t < 4; ++t) bC[t] = bN[t];
  }

  int nloc = lane & 15, half = lane >> 4;
#pragma unroll
  for (int rb = 0; rb < 2; ++rb) {
#pragma unroll
    for (int t = 0; t < 4; ++t) {
      int o = o0 + t * 16 + nloc;
      int s = o / C_;
      int rem = o - s * C_;
      int h = rem >> 6, d = rem & 63;
#pragma unroll
      for (int r = 0; r < 8; ++r) {
        int row = r0 + rb * 16 + r + 8 * half;  // D-layout: vgpr r is row r+8*half
        int bI = row >> 11, n = row & (N_ - 1);
        size_t head = (size_t)(bI * H_ + h) * (size_t)(N_ * D_);
        float val = acc[rb][t][r];
        if (s == 0)      Qb[head + (size_t)n * D_ + d]  = f2bf(val * SCALE_);
        else if (s == 1) Kb[head + (size_t)n * D_ + d]  = f2bf(val);
        else             Vtb[head + (size_t)d * N_ + n] = f2bf(val);
      }
    }
  }
}

// ------------------------------------------------------------- flash attention
// One wave per (b, h, 16-query tile). Online softmax; 8 WMMA per 32-key chunk.
// K fragments for the next chunk and V fragments for this chunk are loaded
// before the softmax VALU work so loads overlap with math.
__global__ __launch_bounds__(32)
void flash_kernel(const u16* __restrict__ Q, const u16* __restrict__ K,
                  const u16* __restrict__ Vt, u16* __restrict__ attnb) {
  int lane = threadIdx.x;
  int qt = blockIdx.x, h = blockIdx.y, b = blockIdx.z;
  size_t head = (size_t)(b * H_ + h) * (size_t)(N_ * D_);
  const u16* Qh = Q + head;
  const u16* Kh = K + head;
  const u16* Vh = Vt + head;   // Vt layout per head: [D_][N_]
  int q0 = qt * 16;
  int half = lane >> 4, nloc = lane & 15;

  v16bf aQ0 = load_a_frag(Qh + (size_t)q0 * D_, D_, lane);       // K-dim 0..31
  v16bf aQ1 = load_a_frag(Qh + (size_t)q0 * D_ + 32, D_, lane);  // K-dim 32..63

  v8f oA[4];
#pragma unroll
  for (int t = 0; t < 4; ++t) oA[t] = vzero8();
  float mrun[8], lrun[8];
#pragma unroll
  for (int r = 0; r < 8; ++r) { mrun[r] = -1e30f; lrun[r] = 0.0f; }

  __shared__ uint32_t pl[256];       // 16 rows x 32 cols of bf16
  u16* pls = (u16*)pl;

  // preload K fragments for chunk 0: [keytile(2)][kchunk(2)]
  v16bf kf[4];
  kf[0] = load_b_frag(Kh, D_, lane);
  kf[1] = load_b_frag(Kh + 32, D_, lane);
  kf[2] = load_b_frag(Kh + 16 * D_, D_, lane);
  kf[3] = load_b_frag(Kh + 16 * D_ + 32, D_, lane);

  for (int j = 0; j < N_; j += 32) {
    int jn = (j + 32) & (N_ - 1);   // wrap: harmless extra loads at tail
    int jp = (j + 64) & (N_ - 1);
    // prefetch two chunks ahead (global_prefetch_b8)
    __builtin_prefetch(Kh + (size_t)jp * D_ + lane * 4, 0, 1);
    __builtin_prefetch(Vh + jp + (size_t)lane * N_, 0, 1);

    // next chunk's K fragments (consumed next iteration)
    const u16* Kn = Kh + (size_t)jn * D_;
    v16bf kn[4];
    kn[0] = load_b_frag(Kn, D_, lane);
    kn[1] = load_b_frag(Kn + 32, D_, lane);
    kn[2] = load_b_frag(Kn + 16 * D_, D_, lane);
    kn[3] = load_b_frag(Kn + 16 * D_ + 32, D_, lane);
    // this chunk's V fragments (consumed after softmax)
    const u16* Vc = Vh + j;
    v16bf vf[4];
#pragma unroll
    for (int t = 0; t < 4; ++t)
      vf[t] = load_b_frag(Vc + (size_t)(t * 16) * N_, N_, lane);

    v8f s0 = vzero8(), s1 = vzero8();
    s0 = wmma_bf16(aQ0, kf[0], s0);           // keys j..j+15
    s0 = wmma_bf16(aQ1, kf[1], s0);
    s1 = wmma_bf16(aQ0, kf[2], s1);           // keys j+16..j+31
    s1 = wmma_bf16(aQ1, kf[3], s1);

    float p0[8], p1[8];
#pragma unroll
    for (int r = 0; r < 8; ++r) {
      float a = s0[r], bb = s1[r];
      float tmax = fmaxf(a, bb);
#pragma unroll
      for (int m = 1; m < 16; m <<= 1) tmax = fmaxf(tmax, __shfl_xor(tmax, m, 16));
      float mn = fmaxf(mrun[r], tmax);
      float sc = __expf(mrun[r] - mn);
      float e0 = __expf(a - mn);
      float e1 = __expf(bb - mn);
      float rs = e0 + e1;
#pragma unroll
      for (int m = 1; m < 16; m <<= 1) rs += __shfl_xor(rs, m, 16);
      lrun[r] = lrun[r] * sc + rs;
      mrun[r] = mn;
      p0[r] = e0; p1[r] = e1;
#pragma unroll
      for (int t = 0; t < 4; ++t) oA[t][r] *= sc;
    }

    // stage P (f32 D-layout) -> LDS as 16x32 bf16, reload as A fragment
#pragma unroll
    for (int r = 0; r < 8; ++r) {
      int row = r + 8 * half;
      pls[row * 32 + nloc]      = f2bf(p0[r]);
      pls[row * 32 + 16 + nloc] = f2bf(p1[r]);
    }
    __syncthreads();
    v16bf aP = load_a_frag(pls, 32, lane);
#pragma unroll
    for (int t = 0; t < 4; ++t)
      oA[t] = wmma_bf16(aP, vf[t], oA[t]);
    __syncthreads();
#pragma unroll
    for (int t = 0; t < 4; ++t) kf[t] = kn[t];
  }

#pragma unroll
  for (int r = 0; r < 8; ++r) {
    float inv = 1.0f / lrun[r];
    int row = q0 + r + 8 * half;
    size_t base = (size_t)(b * N_ + row) * C_ + h * D_;
#pragma unroll
    for (int t = 0; t < 4; ++t)
      attnb[base + t * 16 + nloc] = f2bf(oA[t][r] * inv);
  }
}

// ------------------------------------------------------------- out projection
// One wave -> 32 rows x 64 cols, software-pipelined like the QKV GEMM.
__global__ __launch_bounds__(32)
void proj_gemm_kernel(const u16* __restrict__ attnb, const u16* __restrict__ wb,
                      const float* __restrict__ bias, float* __restrict__ out) {
  int lane = threadIdx.x;
  int o0 = blockIdx.x * 64;
  int r0 = blockIdx.y * 32;
  v8f acc[2][4];
#pragma unroll
  for (int rb = 0; rb < 2; ++rb)
#pragma unroll
    for (int t = 0; t < 4; ++t) acc[rb][t] = vzero8();

  v16bf aC[2], bC[4];
#pragma unroll
  for (int rb = 0; rb < 2; ++rb)
    aC[rb] = load_a_frag(attnb + (size_t)(r0 + rb * 16) * C_, C_, lane);
#pragma unroll
  for (int t = 0; t < 4; ++t)
    bC[t] = load_b_frag(wb + (size_t)(o0 + t * 16) * C_, C_, lane);

  for (int c = 0; c < C_; c += 32) {
    int cn = (c + 32 < C_) ? (c + 32) : 0;
    v16bf aN[2], bN[4];
#pragma unroll
    for (int rb = 0; rb < 2; ++rb)
      aN[rb] = load_a_frag(attnb + (size_t)(r0 + rb * 16) * C_ + cn, C_, lane);
#pragma unroll
    for (int t = 0; t < 4; ++t)
      bN[t] = load_b_frag(wb + (size_t)(o0 + t * 16) * C_ + cn, C_, lane);
#pragma unroll
    for (int rb = 0; rb < 2; ++rb)
#pragma unroll
      for (int t = 0; t < 4; ++t)
        acc[rb][t] = wmma_bf16(aC[rb], bC[t], acc[rb][t]);
#pragma unroll
    for (int rb = 0; rb < 2; ++rb) aC[rb] = aN[rb];
#pragma unroll
    for (int t = 0; t < 4; ++t) bC[t] = bN[t];
  }

  int nloc = lane & 15, half = lane >> 4;
#pragma unroll
  for (int rb = 0; rb < 2; ++rb)
#pragma unroll
    for (int t = 0; t < 4; ++t) {
      float bv = bias[o0 + t * 16 + nloc];
#pragma unroll
      for (int r = 0; r < 8; ++r) {
        int row = r0 + rb * 16 + r + 8 * half;
        out[(size_t)row * C_ + o0 + t * 16 + nloc] = acc[rb][t][r] + bv;
      }
    }
}

// ---------------------------------------------------------------------------
extern "C" void kernel_launch(void* const* d_in, const int* in_sizes, int n_in,
                              void* d_out, int out_size, void* d_ws, size_t ws_size,
                              hipStream_t stream) {
  (void)in_sizes; (void)n_in; (void)out_size; (void)ws_size;
  const float* x      = (const float*)d_in[0];
  const float* w_qkv  = (const float*)d_in[1];
  const float* w_proj = (const float*)d_in[2];
  const float* b_proj = (const float*)d_in[3];
  float* out = (float*)d_out;

  const size_t xbN    = (size_t)B_ * N_ * C_;       // 6.29M
  const size_t wqkvN  = (size_t)3 * C_ * C_;        // 1.77M
  const size_t wprojN = (size_t)C_ * C_;            // 0.59M
  const size_t headN  = (size_t)B_ * H_ * N_ * D_;  // 6.29M

  u16* p = (u16*)d_ws;
  u16* xb     = p; p += xbN;
  u16* wqkvb  = p; p += wqkvN;
  u16* wprojb = p; p += wprojN;
  u16* Qb     = p; p += headN;
  u16* Kb     = p; p += headN;
  u16* Vtb    = p; p += headN;
  u16* attnb  = xb;   // xb dead after QKV GEMM; reuse for attention output

  cvt_f32_bf16<<<dim3((unsigned)((xbN + 255) / 256)), 256, 0, stream>>>(x, xb, (int)xbN);
  cvt_f32_bf16<<<dim3((unsigned)((wqkvN + 255) / 256)), 256, 0, stream>>>(w_qkv, wqkvb, (int)wqkvN);
  cvt_f32_bf16<<<dim3((unsigned)((wprojN + 255) / 256)), 256, 0, stream>>>(w_proj, wprojb, (int)wprojN);

  qkv_gemm_kernel<<<dim3(3 * C_ / 64, B_ * N_ / 32), 32, 0, stream>>>(xb, wqkvb, Qb, Kb, Vtb);
  flash_kernel<<<dim3(N_ / 16, H_, B_), 32, 0, stream>>>(Qb, Kb, Vtb, attnb);
  proj_gemm_kernel<<<dim3(C_ / 64, B_ * N_ / 32), 32, 0, stream>>>(attnb, wprojb, b_proj, out);
}
